// FPModule_14370960573086
// MI455X (gfx1250) — matrix operand order repaired and checked by reading, not compiled, not run on hardware
//
#include <hip/hip_runtime.h>

// ---------------------------------------------------------------------------
// Problem constants (match reference)
// ---------------------------------------------------------------------------
#define NB        16          // batch (clouds)
#define N_COARSE  1024
#define N_FINE    4096
#define C_IN      256
#define C_SKIP    128
#define C0        (C_IN + C_SKIP)   // 384
#define HW        512               // hidden width
#define M_TOTAL   (NB * N_FINE)     // 65536 fine points
#define BN_EPS    1e-5f
#define KNN_EPS   1e-16f

typedef __bf16 bf16_t;
typedef __attribute__((ext_vector_type(16))) __bf16 v16bf;
typedef __attribute__((ext_vector_type(8)))  __bf16 v8bf;
typedef __attribute__((ext_vector_type(8)))  float  v8f;

// fragment assembled from two 16-byte (b128) LDS loads
union bfrag { v16bf v; v8bf h[2]; };

__device__ __forceinline__ bf16_t f2bf(float f) { return (bf16_t)f; }

// ---------------------------------------------------------------------------
// Kernel 1: fp32 -> bf16 elementwise (weights)
// ---------------------------------------------------------------------------
__global__ void f32_to_bf16_kernel(const float* __restrict__ in,
                                   bf16_t* __restrict__ out, int n) {
    int i = blockIdx.x * 256 + threadIdx.x;
    if (i < n) out[i] = f2bf(in[i]);
}

// ---------------------------------------------------------------------------
// Kernel 2: brute-force 3-NN inverse-distance interpolation + concat (bf16 out)
// One block = 256 fine points of one cloud; coarse positions staged in LDS.
// ---------------------------------------------------------------------------
__global__ __launch_bounds__(256) void knn_interp_concat_kernel(
    const float* __restrict__ x,        // [NB*N_COARSE, C_IN]
    const float* __restrict__ pos,      // [NB*N_COARSE, 3]
    const float* __restrict__ x_skip,   // [NB*N_FINE, C_SKIP]
    const float* __restrict__ pos_skip, // [NB*N_FINE, 3]
    bf16_t* __restrict__ concat_out)    // [M_TOTAL, C0] bf16
{
    __shared__ float spos[N_COARSE * 3];   // 12 KB

    const int blk   = blockIdx.x;          // 0..255
    const int cloud = blk >> 4;            // 16 blocks per cloud
    const int fbase = cloud * N_FINE + (blk & 15) * 256;
    const int cbase = cloud * N_COARSE;
    const int t     = threadIdx.x;

    for (int i = t; i < N_COARSE * 3; i += 256)
        spos[i] = pos[(size_t)cbase * 3 + i];
    __syncthreads();

    const int fi = fbase + t;
    const float qx = pos_skip[(size_t)fi * 3 + 0];
    const float qy = pos_skip[(size_t)fi * 3 + 1];
    const float qz = pos_skip[(size_t)fi * 3 + 2];

    float d0 = 1e30f, d1 = 1e30f, d2 = 1e30f;
    int   i0 = 0,     i1 = 0,     i2 = 0;
    for (int j = 0; j < N_COARSE; ++j) {
        float dx = qx - spos[j * 3 + 0];
        float dy = qy - spos[j * 3 + 1];
        float dz = qz - spos[j * 3 + 2];
        float d  = dx * dx + dy * dy + dz * dz;
        if (d < d2) {
            if (d < d1) {
                if (d < d0) { d2 = d1; i2 = i1; d1 = d0; i1 = i0; d0 = d; i0 = j; }
                else        { d2 = d1; i2 = i1; d1 = d;  i1 = j; }
            } else          { d2 = d;  i2 = j; }
        }
    }

    float w0 = 1.0f / fmaxf(d0, KNN_EPS);
    float w1 = 1.0f / fmaxf(d1, KNN_EPS);
    float w2 = 1.0f / fmaxf(d2, KNN_EPS);
    const float inv = 1.0f / (w0 + w1 + w2);
    w0 *= inv; w1 *= inv; w2 *= inv;

    const float* f0 = x + (size_t)(cbase + i0) * C_IN;
    const float* f1 = x + (size_t)(cbase + i1) * C_IN;
    const float* f2 = x + (size_t)(cbase + i2) * C_IN;
    bf16_t* row = concat_out + (size_t)fi * C0;

    #pragma unroll 4
    for (int c = 0; c < C_IN; c += 4) {
        float4 a  = *(const float4*)(f0 + c);
        float4 b  = *(const float4*)(f1 + c);
        float4 cc = *(const float4*)(f2 + c);
        row[c + 0] = f2bf(w0 * a.x + w1 * b.x + w2 * cc.x);
        row[c + 1] = f2bf(w0 * a.y + w1 * b.y + w2 * cc.y);
        row[c + 2] = f2bf(w0 * a.z + w1 * b.z + w2 * cc.z);
        row[c + 3] = f2bf(w0 * a.w + w1 * b.w + w2 * cc.w);
    }
    const float* sk = x_skip + (size_t)fi * C_SKIP;
    #pragma unroll 4
    for (int c = 0; c < C_SKIP; c += 4) {
        float4 s = *(const float4*)(sk + c);
        row[C_IN + c + 0] = f2bf(s.x);
        row[C_IN + c + 1] = f2bf(s.y);
        row[C_IN + c + 2] = f2bf(s.z);
        row[C_IN + c + 3] = f2bf(s.w);
    }
}

// ---------------------------------------------------------------------------
// Kernel 3: bf16 WMMA GEMM  D = relu(A @ W + bias) -> BatchNorm (inference)
// Block tile 128(M) x 64(N), 8 waves; wave tile 32x32 = 2x2 WMMA 16x16 tiles.
// K stepped by 64 through LDS (2 WMMA sub-steps per barrier).
// A:[M,K] bf16 row-major.  W:[K,N] bf16 row-major, stored TRANSPOSED in LDS
// so every B fragment is two runs of 8 contiguous bf16 (2x ds_load_b128, no repacking).
// ---------------------------------------------------------------------------
__global__ __launch_bounds__(256) void gemm_relu_bn_kernel(
    const bf16_t* __restrict__ A, const bf16_t* __restrict__ W,
    const float* __restrict__ bias, const float* __restrict__ gamma,
    const float* __restrict__ beta, const float* __restrict__ mean,
    const float* __restrict__ var,
    bf16_t* __restrict__ out_bf, float* __restrict__ out_f32,
    int N, int K)
{
    __shared__ bf16_t As [128][72];  // [m][k] 64 K used + 8 pad (144B rows, 16B-aligned)
    __shared__ bf16_t Bst[64][72];   // [n][k] transposed W tile, 64 K used + 8 pad

    const int tid  = threadIdx.x;
    const int wave = tid >> 5;
    const int lane = tid & 31;
    const int lm   = lane & 15;      // row (A) / col (B/D) within 16
    const int kh   = lane >> 4;      // K-half selector per ISA 16-bit layouts
    const int wm   = wave & 3;       // wave M sub-tile (4 x 32 rows)
    const int wn   = wave >> 2;      // wave N sub-tile (2 x 32 cols)
    const int m0   = blockIdx.x * 128;
    const int n0   = blockIdx.y * 64;

    v8f acc[2][2];
    #pragma unroll
    for (int mi = 0; mi < 2; ++mi)
        #pragma unroll
        for (int ni = 0; ni < 2; ++ni)
            #pragma unroll
            for (int r = 0; r < 8; ++r) acc[mi][ni][r] = 0.0f;

    // cooperative tile-load assignments
    const int ar   = tid >> 1;          // A row 0..127
    const int aseg = (tid & 1) * 32;    // 32 bf16 per thread

    for (int k0 = 0; k0 < K; k0 += 64) {
        // ---- A tile: 128 x 64 bf16, 4x uint4 per thread, direct layout
        const bf16_t* ga = A + (size_t)(m0 + ar) * K + k0 + aseg;
        #pragma unroll
        for (int q = 0; q < 4; ++q)
            *(uint4*)&As[ar][aseg + q * 8] = *(const uint4*)(ga + q * 8);

        // ---- W tile: 64(k) x 64(n) bf16, transposed into Bst[n][k]
        #pragma unroll
        for (int q = 0; q < 2; ++q) {
            const int lin  = tid + q * 256;
            const int kr   = lin >> 3;          // 0..63
            const int nseg = (lin & 7) * 8;     // 0..56
            uint4 wv = *(const uint4*)(W + (size_t)(k0 + kr) * N + n0 + nseg);
            const bf16_t* wp = (const bf16_t*)&wv;
            #pragma unroll
            for (int e = 0; e < 8; ++e)
                Bst[nseg + e][kr] = wp[e];
        }

        if (k0 + 64 < K) {                       // gfx1250 global_prefetch_b8
            __builtin_prefetch(ga + 64, 0, 3);
            __builtin_prefetch(W + (size_t)(k0 + 64) * N + n0, 0, 3);
        }
        __syncthreads();

        // ---- 2 K-sub-steps of 32, 4 WMMA each
        #pragma unroll
        for (int ks = 0; ks < 2; ++ks) {
            bfrag afr[2], bfr[2];
            #pragma unroll
            for (int mi = 0; mi < 2; ++mi) {
                const bf16_t* ap = &As[wm * 32 + mi * 16 + lm][ks * 32];
                afr[mi].h[0] = *(const v8bf*)(ap + kh * 8);        // K = kh*8 + 0..7
                afr[mi].h[1] = *(const v8bf*)(ap + 16 + kh * 8);   // K = 16 + kh*8 + 0..7
            }
            #pragma unroll
            for (int ni = 0; ni < 2; ++ni) {
                const bf16_t* bp = &Bst[wn * 32 + ni * 16 + lm][ks * 32];
                bfr[ni].h[0] = *(const v8bf*)(bp + kh * 8);        // K = kh*8 + 0..7
                bfr[ni].h[1] = *(const v8bf*)(bp + 16 + kh * 8);   // K = 16 + kh*8 + 0..7
            }
            #pragma unroll
            for (int mi = 0; mi < 2; ++mi)
                #pragma unroll
                for (int ni = 0; ni < 2; ++ni)
                    acc[mi][ni] = __builtin_amdgcn_wmma_f32_16x16x32_bf16(
                        false, afr[mi].v, false, bfr[ni].v,
                        (short)0, acc[mi][ni], false, false);
        }
        __syncthreads();
    }

    // Fused bias + ReLU + BatchNorm epilogue.
    // D layout: lanes 0-15: N=lm, M=r; lanes 16-31: N=lm, M=8+r.
    #pragma unroll
    for (int ni = 0; ni < 2; ++ni) {
        const int col = n0 + wn * 32 + ni * 16 + lm;
        const float sc = gamma[col] * rsqrtf(var[col] + BN_EPS);
        const float mn = mean[col];
        const float bt = beta[col];
        const float bs = bias[col];
        #pragma unroll
        for (int mi = 0; mi < 2; ++mi) {
            const int rowb = m0 + wm * 32 + mi * 16 + kh * 8;
            #pragma unroll
            for (int r = 0; r < 8; ++r) {
                float h = fmaxf(acc[mi][ni][r] + bs, 0.0f);
                float y = (h - mn) * sc + bt;
                const size_t idx = (size_t)(rowb + r) * N + col;
                if (out_bf)  out_bf[idx]  = f2bf(y);
                if (out_f32) out_f32[idx] = y;
            }
        }
    }
}

// ---------------------------------------------------------------------------
// Kernel 4: copy pos_skip + batch_skip into d_out tail
// ---------------------------------------------------------------------------
__global__ void tail_copy_kernel(const float* __restrict__ pos_skip,
                                 const int* __restrict__ batch_skip,
                                 float* __restrict__ out_tail)
{
    const int i = blockIdx.x * 256 + threadIdx.x;
    const int npos = M_TOTAL * 3;
    if (i < npos) {
        out_tail[i] = pos_skip[i];
    } else {
        const int j = i - npos;
        if (j < M_TOTAL) out_tail[npos + j] = (float)batch_skip[j];
    }
}

// ---------------------------------------------------------------------------
// Host launcher
// ---------------------------------------------------------------------------
extern "C" void kernel_launch(void* const* d_in, const int* in_sizes, int n_in,
                              void* d_out, int out_size, void* d_ws, size_t ws_size,
                              hipStream_t stream) {
    (void)in_sizes; (void)n_in; (void)out_size; (void)ws_size;

    const float* x         = (const float*)d_in[0];
    const float* pos       = (const float*)d_in[1];
    /* d_in[2] = batch (coarse) unused: contiguous equal-size clouds */
    const float* x_skip    = (const float*)d_in[3];
    const float* pos_skip  = (const float*)d_in[4];
    const int*   batch_sk  = (const int*)d_in[5];
    const float* w1 = (const float*)d_in[6];
    const float* b1 = (const float*)d_in[7];
    const float* g1 = (const float*)d_in[8];
    const float* be1= (const float*)d_in[9];
    const float* m1 = (const float*)d_in[10];
    const float* v1 = (const float*)d_in[11];
    const float* w2 = (const float*)d_in[12];
    const float* b2 = (const float*)d_in[13];
    const float* g2 = (const float*)d_in[14];
    const float* be2= (const float*)d_in[15];
    const float* m2 = (const float*)d_in[16];
    const float* v2 = (const float*)d_in[17];

    // Workspace layout (bf16 buffers)
    char* ws = (char*)d_ws;
    bf16_t* concat_bf = (bf16_t*)ws;                                    // M x 384
    size_t off = (size_t)M_TOTAL * C0 * sizeof(bf16_t);
    bf16_t* h1_bf = (bf16_t*)(ws + off);                                // M x 512
    off += (size_t)M_TOTAL * HW * sizeof(bf16_t);
    bf16_t* w1_bf = (bf16_t*)(ws + off);                                // 384 x 512
    off += (size_t)C0 * HW * sizeof(bf16_t);
    bf16_t* w2_bf = (bf16_t*)(ws + off);                                // 512 x 512

    // 1) weights -> bf16
    f32_to_bf16_kernel<<<(C0 * HW + 255) / 256, 256, 0, stream>>>(w1, w1_bf, C0 * HW);
    f32_to_bf16_kernel<<<(HW * HW + 255) / 256, 256, 0, stream>>>(w2, w2_bf, HW * HW);

    // 2) knn-interpolate + concat -> bf16 activation matrix
    knn_interp_concat_kernel<<<256, 256, 0, stream>>>(x, pos, x_skip, pos_skip,
                                                      concat_bf);

    // 3) layer 1: [65536,384] @ [384,512] fused relu+bn -> bf16
    dim3 grid(M_TOTAL / 128, HW / 64);
    gemm_relu_bn_kernel<<<grid, 256, 0, stream>>>(
        concat_bf, w1_bf, b1, g1, be1, m1, v1,
        h1_bf, (float*)nullptr, HW, C0);

    // 4) layer 2: [65536,512] @ [512,512] fused relu+bn -> fp32 d_out
    gemm_relu_bn_kernel<<<grid, 256, 0, stream>>>(
        h1_bf, w2_bf, b2, g2, be2, m2, v2,
        (bf16_t*)nullptr, (float*)d_out, HW, HW);

    // 5) tail: pos_skip + batch_skip
    float* tail = (float*)d_out + (size_t)M_TOTAL * HW;
    tail_copy_kernel<<<(M_TOTAL * 4 + 255) / 256, 256, 0, stream>>>(
        pos_skip, batch_sk, tail);
}
